// Attention_21680994910660
// MI455X (gfx1250) — compile-verified, matrix operand order checked by reading, not run on hardware
//
#include <hip/hip_runtime.h>
#include <hip/hip_bf16.h>
#include <cstddef>

// MI455X / gfx1250 attention pipeline, wave32 + v_wmma_f32_16x16x32_bf16.
// B=16, S=1024, D=512 (from reference setup_inputs).

typedef __attribute__((ext_vector_type(16))) __bf16  v16bf;
typedef __attribute__((ext_vector_type(2)))  __bf16  v2bf;
typedef __attribute__((ext_vector_type(8)))  float   v8f;
typedef __attribute__((ext_vector_type(16))) float   v16f;

static constexpr int B_ = 16;
static constexpr int S_ = 1024;
static constexpr int D_ = 512;

// f32 -> bf16 with round-to-nearest-even (bit arithmetic fallback).
__device__ __forceinline__ unsigned short f2bfbits(float f) {
  union { float f; unsigned u; } x; x.f = f;
  unsigned u = x.u;
  u += 0x7FFFu + ((u >> 16) & 1u);
  return (unsigned short)(u >> 16);
}
__device__ __forceinline__ __bf16 f2bf(float f) {
  union { unsigned short s; __bf16 b; } y;
  y.s = f2bfbits(f);
  return y.b;
}

// Convert 16 floats -> 16 bf16 (packed HW converter if the builtin exists).
__device__ __forceinline__ v16bf cvt16(v16f f) {
  v16bf r;
#if __has_builtin(__builtin_amdgcn_cvt_pk_bf16_f32)
#pragma unroll
  for (int e = 0; e < 8; ++e) {
    v2bf p = __builtin_amdgcn_cvt_pk_bf16_f32(f[2 * e], f[2 * e + 1]);
    r[2 * e] = p[0];
    r[2 * e + 1] = p[1];
  }
#else
#pragma unroll
  for (int e = 0; e < 16; ++e) r[e] = f2bf(f[e]);
#endif
  return r;
}

__device__ __forceinline__ v8f wmma_bf16(v16bf a, v16bf b, v8f c) {
  // (neg_a, A, neg_b, B, c_mod, C, reuse_a, reuse_b)
  return __builtin_amdgcn_wmma_f32_16x16x32_bf16(false, a, false, b,
                                                 (short)0, c, false, false);
}

// ---------------------------------------------------------------------------
// Kernel 0: one-time f32 -> bf16 conversion of the three 512x512 weight
// matrices (z selects which). Removes the per-strip redundant conversion of
// W fragments inside k_proj's K-loop (W rows were re-converted per M-tile).
// ---------------------------------------------------------------------------
__global__ __launch_bounds__(256) void k_cvtw(
    const float* __restrict__ WQ, const float* __restrict__ WK,
    const float* __restrict__ WV, unsigned short* __restrict__ wbf) {
  const int z = blockIdx.z;
  const float* W = (z == 0) ? WQ : (z == 1) ? WK : WV;
  unsigned short* O = wbf + (size_t)z * D_ * D_;
  const int i = (blockIdx.x * 256 + (int)threadIdx.x) * 4;  // 0..D*D-4
  const float4 f = *(const float4*)(W + i);
  ushort4 o;
  o.x = f2bfbits(f.x);
  o.y = f2bfbits(f.y);
  o.z = f2bfbits(f.z);
  o.w = f2bfbits(f.w);
  *(ushort4*)(O + i) = o;
}

// ---------------------------------------------------------------------------
// Kernel 1: q = Q·WQᵀ, k = K·WKᵀ, v = V·WVᵀ  (z selects which matrix).
// C[m,n] = Σ_d X[m,d]·W[n,d]; B-operand column n == row n of W (contiguous,
// already bf16 from k_cvtw). A converted f32->bf16 in-loop (streamed once).
// q,k stored row-major bf16; v stored transposed (vt[b][d][s]) for k_out.
// Each wave: one 16x64 output strip; K-loop of 16 WMMA steps (D=512).
// ---------------------------------------------------------------------------
__global__ __launch_bounds__(128) void k_proj(
    const float* __restrict__ Q, const float* __restrict__ K,
    const float* __restrict__ V, const unsigned short* __restrict__ wbf,
    unsigned short* __restrict__ qbf, unsigned short* __restrict__ kbf,
    unsigned short* __restrict__ vtb) {
  const int z = blockIdx.z;
  const float* X = (z == 0) ? Q : (z == 1) ? K : V;
  const unsigned short* W = wbf + (size_t)z * D_ * D_;

  const int wave  = threadIdx.x >> 5;
  const int lane  = threadIdx.x & 31;
  const int strip = blockIdx.x * 4 + wave;   // 0..8191
  const int tm    = strip >> 3;              // M tile 0..1023 (rows of B*S)
  const int tn0   = (strip & 7) << 2;        // first of 4 N tiles (32 total)
  const int h     = lane >> 4;               // lane half: K 0-15 vs 16-31
  const int m     = lane & 15;

  v8f c[4] = {};
  const int arow = tm * 16 + m;
  const float* aptr = X + (size_t)arow * D_ + h * 16;
  for (int k0 = 0; k0 < D_; k0 += 32) {
    // prefetch next K-step of the A stream (global_prefetch_b8)
    __builtin_prefetch(aptr + k0 + 32, 0, 1);
    v16bf a = cvt16(*(const v16f*)(aptr + k0));
#pragma unroll
    for (int t = 0; t < 4; ++t) {
      const int n = (tn0 + t) * 16 + m;
      v16bf b = *(const v16bf*)(W + (size_t)n * D_ + k0 + h * 16);
      c[t] = wmma_bf16(a, b, c[t]);
    }
  }

  if (z < 2) {
    unsigned short* O = (z == 0) ? qbf : kbf;
#pragma unroll
    for (int t = 0; t < 4; ++t) {
      const int n = (tn0 + t) * 16 + m;
#pragma unroll
      for (int j = 0; j < 8; ++j) {
        const int r = tm * 16 + j + 8 * h;          // C layout: M=j+8h, N=m
        O[(size_t)r * D_ + n] = f2bfbits(c[t][j]);
      }
    }
  } else {
#pragma unroll
    for (int t = 0; t < 4; ++t) {
      const int dd = (tn0 + t) * 16 + m;
#pragma unroll
      for (int j = 0; j < 8; ++j) {
        const int r  = tm * 16 + j + 8 * h;          // global row = b*S + s
        const int bb = r >> 10, s = r & (S_ - 1);
        vtb[((size_t)bb * D_ + dd) * S_ + s] = f2bfbits(c[t][j]);
      }
    }
  }
}

// ---------------------------------------------------------------------------
// Kernel 2: scores[b] = q[b] · k[b]ᵀ * (1/sqrt(D)), f32 into attn region.
// B-operand column n == row n of k (contiguous bf16).
// ---------------------------------------------------------------------------
__global__ __launch_bounds__(256) void k_scores(
    const unsigned short* __restrict__ qbf,
    const unsigned short* __restrict__ kbf,
    float* __restrict__ scores) {
  const int b     = blockIdx.y;
  const int wave  = threadIdx.x >> 5;
  const int lane  = threadIdx.x & 31;
  const int strip = blockIdx.x * 8 + wave;   // 0..1023
  const int tm    = strip >> 4;              // 0..63
  const int tn0   = (strip & 15) << 2;       // 64 N tiles / 4
  const int h     = lane >> 4, m = lane & 15;

  const unsigned short* qb = qbf + (size_t)b * S_ * D_;
  const unsigned short* kb = kbf + (size_t)b * S_ * D_;

  v8f c[4] = {};
  const int arow = tm * 16 + m;
  const unsigned short* aptr = qb + (size_t)arow * D_ + h * 16;
  for (int k0 = 0; k0 < D_; k0 += 32) {
    __builtin_prefetch(aptr + k0 + 32, 0, 1);
    v16bf a = *(const v16bf*)(aptr + k0);
#pragma unroll
    for (int t = 0; t < 4; ++t) {
      const int n = (tn0 + t) * 16 + m;
      v16bf bm = *(const v16bf*)(kb + (size_t)n * D_ + k0 + h * 16);
      c[t] = wmma_bf16(a, bm, c[t]);
    }
  }

  const float scale = 0.044194173824159216f;  // 1/sqrt(512)
  float* sb = scores + (size_t)b * S_ * S_;
#pragma unroll
  for (int t = 0; t < 4; ++t) {
    const int n = (tn0 + t) * 16 + m;
#pragma unroll
    for (int j = 0; j < 8; ++j) {
      const int r = tm * 16 + j + 8 * h;
      sb[(size_t)r * S_ + n] = c[t][j] * scale;
    }
  }
}

// ---------------------------------------------------------------------------
// Kernel 3: row softmax + POST-softmax mask overwrite (reference quirk:
// masked entries become -1e9 and participate in the output GEMM).
// One 256-thread block per row (4 elements/thread over S=1024).
// Writes f32 attn output + bf16 copy for the attn·v WMMA GEMM.
// ---------------------------------------------------------------------------
__global__ __launch_bounds__(256) void k_softmax(
    float* __restrict__ attn, const int* __restrict__ mask,
    unsigned short* __restrict__ abf) {
  __shared__ float red[256];
  const int row = blockIdx.x;       // 0..B*S-1
  const int b   = row >> 10;
  const int t   = threadIdx.x;
  float* rp = attn + (size_t)row * S_;

  float v[4], mx = -3.402823466e38f;
#pragma unroll
  for (int i = 0; i < 4; ++i) {
    v[i] = rp[t + i * 256];
    mx = fmaxf(mx, v[i]);
  }
  red[t] = mx; __syncthreads();
  for (int o = 128; o > 0; o >>= 1) {
    if (t < o) red[t] = fmaxf(red[t], red[t + o]);
    __syncthreads();
  }
  const float smax = red[0]; __syncthreads();

  float s = 0.f;
#pragma unroll
  for (int i = 0; i < 4; ++i) { v[i] = __expf(v[i] - smax); s += v[i]; }
  red[t] = s; __syncthreads();
  for (int o = 128; o > 0; o >>= 1) {
    if (t < o) red[t] += red[t + o];
    __syncthreads();
  }
  const float inv = 1.f / red[0];

#pragma unroll
  for (int i = 0; i < 4; ++i) {
    const int k = t + i * 256;
    const float fin = (mask[b * S_ + k] == 0) ? -1.0e9f : v[i] * inv;
    rp[k] = fin;
    abf[(size_t)row * S_ + k] = f2bfbits(fin);
  }
}

// ---------------------------------------------------------------------------
// Kernel 4: out[b] = attn[b] · v[b].  B-operand column n (= feature dd) is a
// contiguous row of vt[b][dd][*].  K-loop over S=1024 (32 WMMA steps).
// ---------------------------------------------------------------------------
__global__ __launch_bounds__(256) void k_out(
    const unsigned short* __restrict__ abf,
    const unsigned short* __restrict__ vtb,
    float* __restrict__ out) {
  const int b     = blockIdx.y;
  const int wave  = threadIdx.x >> 5;
  const int lane  = threadIdx.x & 31;
  const int strip = blockIdx.x * 8 + wave;  // 0..511
  const int tm    = strip >> 3;             // 0..63
  const int tn0   = (strip & 7) << 2;       // 32 N tiles / 4
  const int h     = lane >> 4, m = lane & 15;

  const unsigned short* ab = abf + (size_t)b * S_ * S_;
  const unsigned short* vb = vtb + (size_t)b * D_ * S_;

  v8f c[4] = {};
  const int arow = tm * 16 + m;
  const unsigned short* aptr = ab + (size_t)arow * S_ + h * 16;
  for (int k0 = 0; k0 < S_; k0 += 32) {
    __builtin_prefetch(aptr + k0 + 32, 0, 1);
    v16bf a = *(const v16bf*)(aptr + k0);
#pragma unroll
    for (int t = 0; t < 4; ++t) {
      const int n = (tn0 + t) * 16 + m;
      v16bf bm = *(const v16bf*)(vb + (size_t)n * S_ + k0 + h * 16);
      c[t] = wmma_bf16(a, bm, c[t]);
    }
  }

  float* ob = out + (size_t)b * S_ * D_;
#pragma unroll
  for (int t = 0; t < 4; ++t) {
    const int n = (tn0 + t) * 16 + m;
#pragma unroll
    for (int j = 0; j < 8; ++j) {
      const int r = tm * 16 + j + 8 * h;
      ob[(size_t)r * D_ + n] = c[t][j];
    }
  }
}

// ---------------------------------------------------------------------------
extern "C" void kernel_launch(void* const* d_in, const int* in_sizes, int n_in,
                              void* d_out, int out_size, void* d_ws,
                              size_t ws_size, hipStream_t stream) {
  (void)in_sizes; (void)n_in; (void)out_size; (void)ws_size;

  const float* Q    = (const float*)d_in[0];
  const float* K    = (const float*)d_in[1];
  const float* V    = (const float*)d_in[2];
  const float* WQ   = (const float*)d_in[3];
  const float* WK   = (const float*)d_in[4];
  const float* WV   = (const float*)d_in[5];
  const int*   mask = (const int*)d_in[6];

  float* out  = (float*)d_out;
  float* attn = out + (size_t)B_ * S_ * D_;   // outputs concatenated flat

  // Workspace (~81.5 MB): bf16 q, k, v-transposed (16 MB each),
  // bf16 attn (32 MB), bf16 weights (1.5 MB).
  unsigned short* qbf = (unsigned short*)d_ws;
  unsigned short* kbf = qbf + (size_t)B_ * S_ * D_;
  unsigned short* vtb = kbf + (size_t)B_ * S_ * D_;
  unsigned short* abf = vtb + (size_t)B_ * S_ * D_;
  unsigned short* wbf = abf + (size_t)B_ * S_ * S_;

  // 0) convert weights to bf16 once (removes in-loop W conversion in k_proj)
  k_cvtw<<<dim3(D_ * D_ / 1024, 1, 3), 256, 0, stream>>>(WQ, WK, WV, wbf);
  // 1) projections: 3 matrices x 8192 strips / 4 waves per block
  k_proj<<<dim3(2048, 1, 3), 128, 0, stream>>>(Q, K, V, wbf, qbf, kbf, vtb);
  // 2) scores into attn region: per batch 1024 strips / 8 waves
  k_scores<<<dim3(128, B_), 256, 0, stream>>>(qbf, kbf, attn);
  // 3) softmax + post-softmax mask, one block per row
  k_softmax<<<dim3(B_ * S_), 256, 0, stream>>>(attn, mask, abf);
  // 4) attn·v: per batch 512 strips / 8 waves
  k_out<<<dim3(64, B_), 256, 0, stream>>>(abf, vtb, out);
}
